// GraphAttentionLayer_85349590106564
// MI455X (gfx1250) — matrix operand order repaired and checked by reading, not compiled, not run on hardware
//
#include <hip/hip_runtime.h>
#include <hip/hip_bf16.h>

// ---------------------------------------------------------------------------
// GAT layer, N=4096, F_IN=128, H=4, D=64.
// Fused masked-softmax + aggregation with v_wmma_f32_16x16x32_f16.
// Scores are rank-1: e_ij = leaky(el_i + er_j). leaky is monotone, so the
// exact unmasked row max is M_i = leaky(el_i + max_j er_j): a fixed softmax
// stabilizer -> single-pass softmax, no running max / accumulator rescale.
// ---------------------------------------------------------------------------

#define N_NODES 4096
#define F_IN    128
#define N_HEADS 4
#define N_HID   64
#define F_OUT   256   // N_HEADS * N_HID
#define NEG_SLOPE 0.2f
#define LOG2E   1.442695040888963f

typedef _Float16 v16h __attribute__((ext_vector_type(16)));
typedef float    v8f  __attribute__((ext_vector_type(8)));
typedef float    vf4  __attribute__((ext_vector_type(4)));

// ---------------------------------------------------------------------------
// Kernel 0: fold w_attn into W_proj_attn.
//   v[side][h][f] = sum_k W_proj_attn[f, h*32+k] * w_attn[side*32+k]
// ---------------------------------------------------------------------------
__global__ void gat_attnvec_kernel(const float* __restrict__ Wpa,   // [128][128]
                                   const float* __restrict__ wattn, // [64]
                                   float* __restrict__ v)           // [2][4][128]
{
    int idx = blockIdx.x * blockDim.x + threadIdx.x;  // 4 blocks * 256
    if (idx >= 2 * N_HEADS * F_IN) return;
    int side = idx >> 9;
    int hh   = (idx >> 7) & 3;
    int f    = idx & 127;
    float acc = 0.f;
#pragma unroll
    for (int k = 0; k < 32; ++k)
        acc = fmaf(Wpa[f * (N_HEADS * 32) + hh * 32 + k], wattn[side * 32 + k], acc);
    v[idx] = acc;
}

// ---------------------------------------------------------------------------
// Kernel 1: value projection, stored TRANSPOSED in f16:
//   gT[col][n] = f16( sum_f h[n,f] * W_proj[f,col] ),  col = head*64+d
// ---------------------------------------------------------------------------
__global__ void gat_proj_kernel(const float* __restrict__ h,   // [4096][128]
                                const float* __restrict__ W,   // [128][256]
                                _Float16* __restrict__ gT)     // [256][4096]
{
    __shared__ float hrow[F_IN];
    int n   = blockIdx.x;
    int col = threadIdx.x;                 // 0..255
    if (col < F_IN) hrow[col] = h[n * F_IN + col];
    __syncthreads();
    float acc = 0.f;
#pragma unroll 8
    for (int f = 0; f < F_IN; ++f)
        acc = fmaf(hrow[f], W[f * F_OUT + col], acc);
    gT[(size_t)col * N_NODES + n] = (_Float16)acc;
}

// ---------------------------------------------------------------------------
// Kernel 2: el[h][n], er[h][n]  (128-wide dot per node per head per side)
// ---------------------------------------------------------------------------
__global__ void gat_eler_kernel(const float* __restrict__ h,  // [4096][128]
                                const float* __restrict__ v,  // [2][4][128]
                                float* __restrict__ el,       // [4][4096]
                                float* __restrict__ er)       // [4][4096]
{
    int tid  = threadIdx.x;                       // 512 threads
    int n    = blockIdx.x * 64 + (tid & 63);
    int hs   = tid >> 6;                          // 0..7
    int side = hs >> 2, hh = hs & 3;
    const float* vp = v + side * (N_HEADS * F_IN) + hh * F_IN;
    const float* hp = h + (size_t)n * F_IN;
    float acc = 0.f;
#pragma unroll 8
    for (int f = 0; f < F_IN; ++f) acc = fmaf(hp[f], vp[f], acc);
    (side ? er : el)[hh * N_NODES + n] = acc;
}

// ---------------------------------------------------------------------------
// Kernel 3: er_max[h] = max_n er[h][n]   (softmax stabilizer per head)
// ---------------------------------------------------------------------------
__global__ void gat_ermax_kernel(const float* __restrict__ er,  // [4][4096]
                                 float* __restrict__ ermax)     // [4]
{
    __shared__ float red[256];
    int head = blockIdx.x;
    int tid  = threadIdx.x;
    float m = -1e30f;
    for (int n = tid; n < N_NODES; n += 256)
        m = fmaxf(m, er[head * N_NODES + n]);
    red[tid] = m;
    __syncthreads();
#pragma unroll
    for (int s = 128; s > 0; s >>= 1) {
        if (tid < s) red[tid] = fmaxf(red[tid], red[tid + s]);
        __syncthreads();
    }
    if (tid == 0) ermax[head] = red[0];
}

// ---------------------------------------------------------------------------
// Main kernel: one wave32 per (16-row i-tile, head). Single-pass softmax with
// fixed per-row max, fused with out += P @ G via 4x v_wmma_f32_16x16x32_f16
// per 32-j chunk (D=64 -> four 16-col tiles).
//
// Per element the probability is computed branch-free:
//   arg = (leaky(x) - Mi) * log2(e) = fma(x, slope_sel, -Mi*log2e)
//   p   = mask ? exp2(arg) : 0        (exp unconditional -> v_cndmask, no
//                                      EXEC-divergent predication)
//
// WMMA wave32 layouts (16-bit operands):
//   A (16x32):  lane L holds row m=L%16; v16h elem t -> k = hi*16 + t,
//               hi = L>=16.
//   B (32x16):  lane L holds col n=L%16; v16h elem j -> k = hi*16 + j
//               => 16 contiguous halves of a gT row, one 32B load.
//   C/D (16x16 f32): lane L, VGPR r -> row M = r + 8*hi, col = L%16.
// ---------------------------------------------------------------------------
__global__ __launch_bounds__(32) void gat_flash_kernel(
    const float*    __restrict__ adj,   // [4096][4096]
    const _Float16* __restrict__ gT,    // [256][4096]
    const float*    __restrict__ el,    // [4][4096]
    const float*    __restrict__ er,    // [4][4096]
    const float*    __restrict__ ermax, // [4]
    float*          __restrict__ out)   // [4096][256]
{
    const int N    = N_NODES;
    const int lane = threadIdx.x;
    const int m    = lane & 15;
    const int hi   = lane >> 4;        // 0 | 1
    const int i0   = blockIdx.x * 16;
    const int head = blockIdx.y;
    const int row  = i0 + m;

    const float el_i = el[head * N + row];
    const float xm   = el_i + ermax[head];
    const float Mi   = xm > 0.f ? xm : NEG_SLOPE * xm;
    const float nMiC = -Mi * LOG2E;            // folded into the exp2 argument

    const float* adj_row = adj + (size_t)row * N + hi * 16;
    const float* er_h    = er + head * N + hi * 16;
    const _Float16* gT_h = gT + (size_t)(head * N_HID + m) * N + hi * 16;

    vf4 psum = {0.f, 0.f, 0.f, 0.f};           // 4 independent chains
    v8f acc0 = {}, acc1 = {}, acc2 = {}, acc3 = {};

#pragma unroll 2
    for (int j0 = 0; j0 < N; j0 += 32) {
        // speculative prefetch of next chunk (silently dropped if invalid)
        __builtin_prefetch(adj_row + j0 + 32, 0, 1);

        // ---- load adjacency + er in A-operand element order ----
        vf4 a0 = *(const vf4*)(adj_row + j0);
        vf4 a1 = *(const vf4*)(adj_row + j0 + 4);
        vf4 a2 = *(const vf4*)(adj_row + j0 + 8);
        vf4 a3 = *(const vf4*)(adj_row + j0 + 12);
        vf4 e0 = *(const vf4*)(er_h + j0);
        vf4 e1 = *(const vf4*)(er_h + j0 + 4);
        vf4 e2 = *(const vf4*)(er_h + j0 + 8);
        vf4 e3 = *(const vf4*)(er_h + j0 + 12);

        v16h Amat;
#pragma unroll
        for (int c = 0; c < 4; ++c) {
            vf4 aq[4] = {a0, a1, a2, a3};
            vf4 eq[4] = {e0, e1, e2, e3};
#pragma unroll
            for (int q = 0; q < 4; ++q) {
                float x     = el_i + eq[q][c];
                float slope = x > 0.f ? LOG2E : (NEG_SLOPE * LOG2E);
                float arg   = fmaf(x, slope, nMiC);          // (leaky(x)-Mi)*log2e
                float pe    = __builtin_amdgcn_exp2f(arg);   // unconditional exp
                pe = (aq[q][c] >= 0.5f) ? pe : 0.f;          // mask via cndmask
                psum[q] += pe;                               // 4 parallel chains
                Amat[q * 4 + c] = (_Float16)pe;
            }
        }

        // ---- B tiles: 16 contiguous halves of 4 gT rows ----
        const _Float16* gp = gT_h + j0;
        v16h B0 = *(const v16h*)(gp + (size_t)0 * 16 * N);
        v16h B1 = *(const v16h*)(gp + (size_t)1 * 16 * N);
        v16h B2 = *(const v16h*)(gp + (size_t)2 * 16 * N);
        v16h B3 = *(const v16h*)(gp + (size_t)3 * 16 * N);

        acc0 = __builtin_amdgcn_wmma_f32_16x16x32_f16(false, Amat, false, B0,
                                                      (short)0, acc0, false, false);
        acc1 = __builtin_amdgcn_wmma_f32_16x16x32_f16(false, Amat, false, B1,
                                                      (short)0, acc1, false, false);
        acc2 = __builtin_amdgcn_wmma_f32_16x16x32_f16(false, Amat, false, B2,
                                                      (short)0, acc2, false, false);
        acc3 = __builtin_amdgcn_wmma_f32_16x16x32_f16(false, Amat, false, B3,
                                                      (short)0, acc3, false, false);
    }

    // ---- one-time denominator combine + epilogue ----
    float total = (psum[0] + psum[1]) + (psum[2] + psum[3]);
    total += __shfl_xor(total, 16, 32);     // lanes m / m+16 hold row m's halves
    float inv = 1.0f / total;
#pragma unroll
    for (int r = 0; r < 8; ++r) {
        float dr = __shfl(inv, r + 8 * hi, 32);   // denom of C-layout row
        int orow = i0 + r + 8 * hi;
        float* op = out + (size_t)orow * F_OUT + head * N_HID + m;
        op[0]  = acc0[r] * dr;
        op[16] = acc1[r] * dr;
        op[32] = acc2[r] * dr;
        op[48] = acc3[r] * dr;
    }
}

// ---------------------------------------------------------------------------
extern "C" void kernel_launch(void* const* d_in, const int* in_sizes, int n_in,
                              void* d_out, int out_size, void* d_ws, size_t ws_size,
                              hipStream_t stream)
{
    (void)in_sizes; (void)n_in; (void)out_size; (void)ws_size;

    const float* h    = (const float*)d_in[0];   // [4096][128]
    const float* adj  = (const float*)d_in[1];   // [4096][4096]
    const float* Wp   = (const float*)d_in[2];   // [128][256]
    const float* Wpa  = (const float*)d_in[3];   // [128][128]
    const float* wat  = (const float*)d_in[4];   // [64]
    float* out = (float*)d_out;                  // [4096][256]

    // workspace layout
    char* ws = (char*)d_ws;
    _Float16* gT = (_Float16*)ws;                              // 256*4096*2 = 2 MiB
    float* el    = (float*)(ws + (size_t)F_OUT * N_NODES * 2); // 4*4096 f32
    float* er    = el + N_HEADS * N_NODES;                     // 4*4096 f32
    float* v     = er + N_HEADS * N_NODES;                     // 2*4*128 f32
    float* ermax = v + 2 * N_HEADS * F_IN;                     // 4 f32

    gat_attnvec_kernel<<<4, 256, 0, stream>>>(Wpa, wat, v);
    gat_proj_kernel<<<N_NODES, 256, 0, stream>>>(h, Wp, gT);
    gat_eler_kernel<<<N_NODES / 64, 512, 0, stream>>>(h, v, el, er);
    gat_ermax_kernel<<<N_HEADS, 256, 0, stream>>>(er, ermax);

    dim3 grid(N_NODES / 16, N_HEADS);
    gat_flash_kernel<<<grid, 32, 0, stream>>>(adj, gT, el, er, ermax, out);
}
